// BertAttention_40544491274671
// MI455X (gfx1250) — matrix operand order repaired and checked by reading, not compiled, not run on hardware
//
#include <hip/hip_runtime.h>
#include <hip/hip_bf16.h>

typedef __attribute__((ext_vector_type(16))) __bf16 v16bf;
typedef __attribute__((ext_vector_type(8)))  float  v8f;

#define HIDDEN 1024
#define NHEADS 16
#define HD     64
#define SEQ    2048
#define BATCH  4
#define MROWS  (BATCH * SEQ) /* 8192 */
#define KBLK   32

// ---- optional CDNA5 async global->LDS path (probe-guarded) ----------------
#if defined(__AMDGCN__) && __has_builtin(__builtin_amdgcn_global_load_async_to_lds_b128)
#define HAVE_ASYNC 1
typedef int v4i __attribute__((vector_size(16)));        // GCC-style int4 (matches builtin proto)
typedef __attribute__((address_space(1))) v4i gv4i;      // global
typedef __attribute__((address_space(3))) v4i lv4i;      // LDS
static __device__ __forceinline__ void async_cp16(const void* g, void* l) {
  __builtin_amdgcn_global_load_async_to_lds_b128((gv4i*)g, (lv4i*)l, 0, 0);
}
static __device__ __forceinline__ void wait_async0() {
#if __has_builtin(__builtin_amdgcn_s_wait_asynccnt)
  __builtin_amdgcn_s_wait_asynccnt(0);
#else
  asm volatile("s_wait_asynccnt 0x0" ::: "memory");
#endif
}
#else
#define HAVE_ASYNC 0
#endif

// ---------------------------------------------------------------------------
// Kernel 1: Y = X * W^T + bias -> bf16 workspace.
// transposeOut==0 : Y[b][h][l][d]   (Q, K)
// transposeOut==1 : Y[b][h][d][l]   (V, so PV B-fragments are contiguous)
// Block = 256 thr (8 waves, 4M x 2N); tile 64(M) x 128(N); K-step 32;
// double-buffered LDS, one barrier per K-step.
// ---------------------------------------------------------------------------
__global__ __launch_bounds__(256) void qkv_proj_kernel(
    const float* __restrict__ X,     // [8192][1024]
    const float* __restrict__ W,     // [1024][1024], row = output column
    const float* __restrict__ bias,  // [1024]
    __bf16* __restrict__ Y,
    int transposeOut)
{
  __shared__ alignas(16) __bf16 sX[2][64][32];
  __shared__ alignas(16) __bf16 sW[2][128][32];

  const int tid  = threadIdx.x;
  const int lane = tid & 31;
  const int w    = tid >> 5;
  const int wm   = w & 3;
  const int wn   = w >> 2;
  const int m0   = blockIdx.x * 64;
  const int n0   = blockIdx.y * 128;
  const int ln   = lane & 15;
  const int kh   = lane >> 4;

  const int xrow = tid >> 2, xc0 = (tid & 3) * 8;   // X: 64x32, 8 floats/thr
  const int wrow = tid >> 1, wc0 = (tid & 1) * 16;  // W: 128x32, 16 floats/thr

  float4 xr0, xr1, wr0, wr1, wr2, wr3;
  auto loadRegs = [&](int k0) {
    const float4* xs = (const float4*)(X + (size_t)(m0 + xrow) * HIDDEN + k0 + xc0);
    xr0 = xs[0]; xr1 = xs[1];
    const float4* ws = (const float4*)(W + (size_t)(n0 + wrow) * HIDDEN + k0 + wc0);
    wr0 = ws[0]; wr1 = ws[1]; wr2 = ws[2]; wr3 = ws[3];
  };
  auto storeTile = [&](int buf) {
    __bf16* dx = &sX[buf][xrow][xc0];
    dx[0] = (__bf16)xr0.x; dx[1] = (__bf16)xr0.y; dx[2] = (__bf16)xr0.z; dx[3] = (__bf16)xr0.w;
    dx[4] = (__bf16)xr1.x; dx[5] = (__bf16)xr1.y; dx[6] = (__bf16)xr1.z; dx[7] = (__bf16)xr1.w;
    __bf16* dw = &sW[buf][wrow][wc0];
    dw[0]  = (__bf16)wr0.x; dw[1]  = (__bf16)wr0.y; dw[2]  = (__bf16)wr0.z; dw[3]  = (__bf16)wr0.w;
    dw[4]  = (__bf16)wr1.x; dw[5]  = (__bf16)wr1.y; dw[6]  = (__bf16)wr1.z; dw[7]  = (__bf16)wr1.w;
    dw[8]  = (__bf16)wr2.x; dw[9]  = (__bf16)wr2.y; dw[10] = (__bf16)wr2.z; dw[11] = (__bf16)wr2.w;
    dw[12] = (__bf16)wr3.x; dw[13] = (__bf16)wr3.y; dw[14] = (__bf16)wr3.z; dw[15] = (__bf16)wr3.w;
  };

  v8f acc[4];
  #pragma unroll
  for (int t = 0; t < 4; ++t) acc[t] = (v8f){0.f,0.f,0.f,0.f,0.f,0.f,0.f,0.f};

  loadRegs(0);
  storeTile(0);

  const int NIT = HIDDEN / KBLK;  // 32
  for (int it = 0; it < NIT; ++it) {
    const int buf = it & 1;
    __syncthreads();                              // buf ready for everyone
    if (it + 1 < NIT) loadRegs((it + 1) * KBLK);  // prefetch next tile

    v16bf a;
    #pragma unroll
    for (int j = 0; j < 8; ++j) {
      const int ke = (j < 4 ? 2*j : 16 + 2*(j - 4)) + kh * 8;
      a[2*j]   = sX[buf][16*wm + ln][ke];
      a[2*j+1] = sX[buf][16*wm + ln][ke + 1];
    }
    #pragma unroll
    for (int t = 0; t < 4; ++t) {
      v16bf bfrag;
      #pragma unroll
      for (int j = 0; j < 8; ++j) {
        const int ke = (j < 4 ? 2*j : 16 + 2*(j - 4)) + kh * 8;
        bfrag[2*j]   = sW[buf][64*wn + 16*t + ln][ke];
        bfrag[2*j+1] = sW[buf][64*wn + 16*t + ln][ke + 1];
      }
      acc[t] = __builtin_amdgcn_wmma_f32_16x16x32_bf16(false, a, false, bfrag,
                                                       (short)0, acc[t], false, false);
    }

    if (it + 1 < NIT) storeTile(buf ^ 1);  // commit prefetch to other buffer
  }

  // epilogue: + bias, scatter bf16
  #pragma unroll
  for (int t = 0; t < 4; ++t) {
    const int gn = n0 + 64*wn + 16*t + ln;
    const float bv = bias[gn];
    const int h = gn >> 6, d = gn & 63;
    #pragma unroll
    for (int r = 0; r < 8; ++r) {
      const int gm = m0 + 16*wm + r + 8*kh;
      const int b = gm >> 11, l = gm & (SEQ - 1);
      const size_t idx = transposeOut
          ? ((size_t)(b * NHEADS + h) * HD + d) * SEQ + l
          : ((size_t)(b * NHEADS + h) * SEQ + l) * HD + d;
      Y[idx] = (__bf16)(acc[t][r] + bv);
    }
  }
}

// ---------------------------------------------------------------------------
// Kernel 2: flash attention.  Block = 256 thr (8 waves), 128 q-rows / block.
// Double-buffered K/V^T staging (async-to-LDS when available), one barrier
// per 32-key block, per-wave LDS transpose of P (no extra barrier).
// ---------------------------------------------------------------------------
__global__ __launch_bounds__(256) void attention_kernel(
    const __bf16* __restrict__ Q,    // [B][H][L][D]
    const __bf16* __restrict__ K,    // [B][H][L][D]
    const __bf16* __restrict__ Vt,   // [B][H][D][L]  (transposed)
    const float*  __restrict__ mask, // [B][L], 1 = attend
    float* __restrict__ out)         // [B][L][HIDDEN]
{
  __shared__ alignas(16) __bf16 sK [2][32][64];
  __shared__ alignas(16) __bf16 sVt[2][64][32];
  __shared__ alignas(16) __bf16 sP[8][16][32];   // per-wave, no cross-wave use

  const int tid  = threadIdx.x;
  const int lane = tid & 31;
  const int w    = tid >> 5;
  const int ln   = lane & 15;
  const int kh   = lane >> 4;
  const int h    = blockIdx.y;
  const int b    = blockIdx.z;
  const int bh   = b * NHEADS + h;
  const int qbase = blockIdx.x * 128 + 16 * w;
  const size_t kBase  = (size_t)bh * SEQ * HD;   // K rows
  const size_t vtBase = (size_t)bh * HD * SEQ;   // V^T rows

  const int krow = tid >> 3, kc0 = (tid & 7) * 8;   // K block 32x64
  const int vrow = tid >> 2, vc0 = (tid & 3) * 8;   // Vt block 64x32

  // Q A-fragments, resident for whole kernel
  v16bf a0, a1;
  {
    const __bf16* qrow = Q + kBase + (size_t)(qbase + ln) * HD;
    #pragma unroll
    for (int j = 0; j < 8; ++j) {
      const int ke = (j < 4 ? 2*j : 16 + 2*(j - 4)) + kh * 8;
      a0[2*j] = qrow[ke];      a0[2*j+1] = qrow[ke + 1];
      a1[2*j] = qrow[32 + ke]; a1[2*j+1] = qrow[32 + ke + 1];
    }
  }

  v8f o[4];
  #pragma unroll
  for (int t = 0; t < 4; ++t) o[t] = (v8f){0.f,0.f,0.f,0.f,0.f,0.f,0.f,0.f};
  float mrow[8], lrow[8];
  #pragma unroll
  for (int r = 0; r < 8; ++r) { mrow[r] = -1e30f; lrow[r] = 0.f; }

#if HAVE_ASYNC
  auto stageAsync = [&](int kb, int buf) {
    async_cp16(K  + kBase  + (size_t)(kb*32 + krow) * HD + kc0, &sK[buf][krow][kc0]);
    async_cp16(Vt + vtBase + (size_t)vrow * SEQ + kb*32 + vc0,  &sVt[buf][vrow][vc0]);
  };
  stageAsync(0, 0);
#else
  uint4 kreg, vreg;
  auto loadRegs = [&](int kb) {
    kreg = *(const uint4*)(K  + kBase  + (size_t)(kb*32 + krow) * HD + kc0);
    vreg = *(const uint4*)(Vt + vtBase + (size_t)vrow * SEQ + kb*32 + vc0);
  };
  auto storeTile = [&](int buf) {
    *(uint4*)&sK[buf][krow][kc0]  = kreg;
    *(uint4*)&sVt[buf][vrow][vc0] = vreg;
  };
  loadRegs(0);
  storeTile(0);
#endif

  const int NKB = SEQ / 32;  // 64
  for (int kb = 0; kb < NKB; ++kb) {
    const int buf = kb & 1;
#if HAVE_ASYNC
    wait_async0();                 // own async stores to LDS done
    __syncthreads();               // everyone's done -> buf valid
    if (kb + 1 < NKB) stageAsync(kb + 1, buf ^ 1);
#else
    __syncthreads();               // buf valid for everyone
    if (kb + 1 < NKB) loadRegs(kb + 1);
#endif

    // ---- S = Q * K^T : two 16x16 key groups, K-chained over d ----
    v8f s[2];
    #pragma unroll
    for (int g = 0; g < 2; ++g) {
      v16bf b0, b1;
      #pragma unroll
      for (int j = 0; j < 8; ++j) {
        const int ke = (j < 4 ? 2*j : 16 + 2*(j - 4)) + kh * 8;
        b0[2*j] = sK[buf][16*g + ln][ke];      b0[2*j+1] = sK[buf][16*g + ln][ke + 1];
        b1[2*j] = sK[buf][16*g + ln][32 + ke]; b1[2*j+1] = sK[buf][16*g + ln][32 + ke + 1];
      }
      v8f z = (v8f){0.f,0.f,0.f,0.f,0.f,0.f,0.f,0.f};
      z = __builtin_amdgcn_wmma_f32_16x16x32_bf16(false, a0, false, b0, (short)0, z, false, false);
      z = __builtin_amdgcn_wmma_f32_16x16x32_bf16(false, a1, false, b1, (short)0, z, false, false);
      s[g] = z;
    }

    // ---- scale + additive mask ----
    const float madd0 = (1.0f - mask[(size_t)b * SEQ + kb*32 + ln])      * -10000.0f;
    const float madd1 = (1.0f - mask[(size_t)b * SEQ + kb*32 + 16 + ln]) * -10000.0f;
    #pragma unroll
    for (int r = 0; r < 8; ++r) {
      s[0][r] = s[0][r] * 0.125f + madd0;
      s[1][r] = s[1][r] * 0.125f + madd1;
    }

    // ---- online softmax (row reductions across 16-lane halves) ----
    float alpha[8];
    #pragma unroll
    for (int r = 0; r < 8; ++r) {
      float mx = fmaxf(s[0][r], s[1][r]);
      mx = fmaxf(mx, __shfl_xor(mx, 1));
      mx = fmaxf(mx, __shfl_xor(mx, 2));
      mx = fmaxf(mx, __shfl_xor(mx, 4));
      mx = fmaxf(mx, __shfl_xor(mx, 8));
      const float mnew = fmaxf(mrow[r], mx);
      alpha[r] = __expf(mrow[r] - mnew);
      mrow[r] = mnew;
      const float p0 = __expf(s[0][r] - mnew);
      const float p1 = __expf(s[1][r] - mnew);
      s[0][r] = p0; s[1][r] = p1;
      float rs = p0 + p1;
      rs += __shfl_xor(rs, 1);
      rs += __shfl_xor(rs, 2);
      rs += __shfl_xor(rs, 4);
      rs += __shfl_xor(rs, 8);
      lrow[r] = lrow[r] * alpha[r] + rs;
    }
    #pragma unroll
    for (int t = 0; t < 4; ++t)
      #pragma unroll
      for (int r = 0; r < 8; ++r) o[t][r] *= alpha[r];

    // ---- P (C layout) -> per-wave LDS -> A layout; same-wave RAW only ----
    #pragma unroll
    for (int r = 0; r < 8; ++r) {
      sP[w][r + 8*kh][ln]      = (__bf16)s[0][r];
      sP[w][r + 8*kh][16 + ln] = (__bf16)s[1][r];
    }
    v16bf pa;
    #pragma unroll
    for (int j = 0; j < 8; ++j) {
      const int ke = (j < 4 ? 2*j : 16 + 2*(j - 4)) + kh * 8;
      pa[2*j]   = sP[w][ln][ke];
      pa[2*j+1] = sP[w][ln][ke + 1];
    }

    // ---- O += P * V : B[k][d] = Vt[d][k] -> contiguous LDS rows ----
    #pragma unroll
    for (int t = 0; t < 4; ++t) {
      v16bf bv;
      #pragma unroll
      for (int j = 0; j < 8; ++j) {
        const int ke = (j < 4 ? 2*j : 16 + 2*(j - 4)) + kh * 8;
        bv[2*j]   = sVt[buf][16*t + ln][ke];
        bv[2*j+1] = sVt[buf][16*t + ln][ke + 1];
      }
      o[t] = __builtin_amdgcn_wmma_f32_16x16x32_bf16(false, pa, false, bv,
                                                     (short)0, o[t], false, false);
    }

#if !HAVE_ASYNC
    if (kb + 1 < NKB) storeTile(buf ^ 1);  // commit prefetch to other buffer
#endif
  }

  // epilogue: divide by l, store [B,L,HIDDEN] fp32
  #pragma unroll
  for (int r = 0; r < 8; ++r) {
    const float inv = 1.0f / lrow[r];
    const int qrow = qbase + r + 8*kh;
    #pragma unroll
    for (int t = 0; t < 4; ++t) {
      out[((size_t)b * SEQ + qrow) * HIDDEN + h * HD + 16*t + ln] = o[t][r] * inv;
    }
  }
}

// ---------------------------------------------------------------------------
extern "C" void kernel_launch(void* const* d_in, const int* in_sizes, int n_in,
                              void* d_out, int out_size, void* d_ws, size_t ws_size,
                              hipStream_t stream) {
  (void)in_sizes; (void)n_in; (void)out_size; (void)ws_size;
  const float* x    = (const float*)d_in[0];
  const float* mask = (const float*)d_in[1];
  const float* Wq   = (const float*)d_in[2];
  const float* bq   = (const float*)d_in[3];
  const float* Wk   = (const float*)d_in[4];
  const float* bk   = (const float*)d_in[5];
  const float* Wv   = (const float*)d_in[6];
  const float* bv   = (const float*)d_in[7];
  float* out = (float*)d_out;

  const size_t per = (size_t)BATCH * NHEADS * SEQ * HD;  // elements per tensor
  __bf16* qws  = (__bf16*)d_ws;
  __bf16* kws  = qws + per;
  __bf16* vtws = kws + per;   // V stored transposed [B,H,D,L]; 48 MB total

  dim3 gProj(MROWS / 64, HIDDEN / 128);
  qkv_proj_kernel<<<gProj, 256, 0, stream>>>(x, Wq, bq, qws, 0);
  qkv_proj_kernel<<<gProj, 256, 0, stream>>>(x, Wk, bk, kws, 0);
  qkv_proj_kernel<<<gProj, 256, 0, stream>>>(x, Wv, bv, vtws, 1);

  dim3 gAtt(SEQ / 128, NHEADS, BATCH);
  attention_kernel<<<gAtt, 256, 0, stream>>>(qws, kws, vtws, mask, out);
}
